// MultiHeadAttention_74285754352267
// MI455X (gfx1250) — matrix operand order repaired and checked by reading, not compiled, not run on hardware
//
#include <hip/hip_runtime.h>
#include <stdint.h>

#define EMB       1024
#define HEADS     16
#define HEAD_DIM  64
#define BATCH     2
#define SEQ       2048
#define MROWS     (BATCH * SEQ)   // 4096

typedef __bf16   v16bf  __attribute__((ext_vector_type(16)));
typedef float    v8f    __attribute__((ext_vector_type(8)));
typedef float    f32x4  __attribute__((ext_vector_type(4)));
typedef uint32_t u32x4  __attribute__((ext_vector_type(4)));
typedef uint32_t u32x2  __attribute__((ext_vector_type(2)));

// WMMA operand fragment: 16 bf16 = 8 VGPRs = 2 x 16-byte loads.
union Frag {
    v16bf    v;
    uint16_t u16[16];
    u32x4    q[2];
};

static __device__ __forceinline__ uint32_t f2bf_bits(float f) {
    uint32_t u = __float_as_uint(f);
    return (u + 0x7FFFu + ((u >> 16) & 1u)) >> 16;  // RNE
}
static __device__ __forceinline__ uint16_t f2bf(float f) {
    return (uint16_t)f2bf_bits(f);
}

// ---------------------------------------------------------------------------
// f32 -> bf16 conversion, 4 elements/thread (float4 in, 2 packed dwords out)
// ---------------------------------------------------------------------------
__global__ __launch_bounds__(256)
void cvt_f32_bf16x4(const float* __restrict__ in,
                    uint16_t* __restrict__ out, int n4) {
    int i = blockIdx.x * blockDim.x + threadIdx.x;
    if (i < n4) {
        f32x4 x = *(const f32x4*)(in + 4 * (size_t)i);
        u32x2 o;
        o[0] = f2bf_bits(x[0]) | (f2bf_bits(x[1]) << 16);
        o[1] = f2bf_bits(x[2]) | (f2bf_bits(x[3]) << 16);
        *(u32x2*)(out + 4 * (size_t)i) = o;
    }
}

// ---------------------------------------------------------------------------
// GEMM: Out[M,N] = (A[M,K] * W[N,K]^T + bias) * oscale
// One wave computes a 32x64 tile: 2 m-subtiles x 4 n-subtiles, K-step 32.
// B fragments shared by both m-subtiles -> 8 WMMAs per 12 b128 loads.
// OUT_TRANS stores Out[col*M + row] (produces V^T for attention).
// oscale folds the attention softmax scale into Q at zero marginal cost.
// ---------------------------------------------------------------------------
template <bool OUT_BF16, bool OUT_TRANS>
__global__ __launch_bounds__(32)
void gemm_xwT(const uint16_t* __restrict__ A,
              const uint16_t* __restrict__ W,
              const float* __restrict__ bias,
              void* __restrict__ Out,
              int Mdim, int Ndim, int Kdim, float oscale) {
    const int lane = threadIdx.x;          // 0..31 (wave32)
    const int hi   = lane >> 4;
    const int lo   = lane & 15;

    const int tiles_n = Ndim >> 6;         // N / 64
    const int tm = blockIdx.x / tiles_n;
    const int tn = blockIdx.x % tiles_n;

    const uint16_t* Ar0 = A + (size_t)(tm * 32 + lo) * Kdim;       // m-sub 0
    const uint16_t* Ar1 = Ar0 + (size_t)16 * Kdim;                 // m-sub 1
    const int achunk = hi * 8;             // A-layout: chunks at c and c+16

    const uint16_t* Wrow[4];
#pragma unroll
    for (int s = 0; s < 4; ++s)
        Wrow[s] = W + (size_t)(tn * 64 + s * 16 + lo) * Kdim;
    const int bchunk = hi * 8;             // B-layout: chunks at c and c+16

    v8f acc0[4], acc1[4];
#pragma unroll
    for (int s = 0; s < 4; ++s)
#pragma unroll
        for (int r = 0; r < 8; ++r) { acc0[s][r] = 0.0f; acc1[s][r] = 0.0f; }

    for (int k0 = 0; k0 < Kdim; k0 += 32) {
        __builtin_prefetch(Ar0 + k0 + 256, 0, 0);   // global_prefetch_b8
        Frag a0, a1;
        a0.q[0] = *(const u32x4*)(Ar0 + k0 + achunk);
        a0.q[1] = *(const u32x4*)(Ar0 + k0 + achunk + 16);
        a1.q[0] = *(const u32x4*)(Ar1 + k0 + achunk);
        a1.q[1] = *(const u32x4*)(Ar1 + k0 + achunk + 16);
#pragma unroll
        for (int s = 0; s < 4; ++s) {
            Frag bfr;
            bfr.q[0] = *(const u32x4*)(Wrow[s] + k0 + bchunk);
            bfr.q[1] = *(const u32x4*)(Wrow[s] + k0 + bchunk + 16);
            acc0[s] = __builtin_amdgcn_wmma_f32_16x16x32_bf16(
                false, a0.v, false, bfr.v, (short)0, acc0[s], false, false);
            acc1[s] = __builtin_amdgcn_wmma_f32_16x16x32_bf16(
                false, a1.v, false, bfr.v, (short)0, acc1[s], false, false);
        }
    }

#pragma unroll
    for (int s = 0; s < 4; ++s) {
        const int col = tn * 64 + s * 16 + lo;
        const float bv = bias[col];
#pragma unroll
        for (int r = 0; r < 8; ++r) {
            const int row0 = tm * 32 + r + 8 * hi;   // C-layout row mapping
            const int row1 = row0 + 16;
            const float v0 = (acc0[s][r] + bv) * oscale;
            const float v1 = (acc1[s][r] + bv) * oscale;
            if (OUT_BF16) {
                uint16_t* o = (uint16_t*)Out;
                if (OUT_TRANS) {
                    o[(size_t)col * Mdim + row0] = f2bf(v0);
                    o[(size_t)col * Mdim + row1] = f2bf(v1);
                } else {
                    o[(size_t)row0 * Ndim + col] = f2bf(v0);
                    o[(size_t)row1 * Ndim + col] = f2bf(v1);
                }
            } else {
                ((float*)Out)[(size_t)row0 * Ndim + col] = v0;
                ((float*)Out)[(size_t)row1 * Ndim + col] = v1;
            }
        }
    }
}

// ---------------------------------------------------------------------------
// Flash attention: one wave per (batch, head, 16 query rows).
// 64 keys/iteration: 8 WMMAs for scores (K fragments contiguous from Kp),
// one online-softmax round per 64 keys (4-way local reduce + 4-step lane
// butterfly), P bounced through LDS into A-layout (16x64), and 8 WMMAs for
// P*V with B-fragments loaded CONTIGUOUSLY from pre-transposed VpT[EMB][M].
// Q is pre-scaled by 1/sqrt(Dh)*log2(e), so scores are used raw in exp2.
// ---------------------------------------------------------------------------
__global__ __launch_bounds__(32)
void attn_fa(const uint16_t* __restrict__ Qp,   // pre-scaled Q
             const uint16_t* __restrict__ Kp,
             const uint16_t* __restrict__ VpT,  // [EMB][MROWS], d-major
             uint16_t* __restrict__ Ctx) {
    __shared__ uint16_t Pl[16 * 64];   // P tile, row-major [q][k]   (2 KB)

    const int lane = threadIdx.x;
    const int hi = lane >> 4, lo = lane & 15;
    const int q0 = blockIdx.x * 16;
    const int h  = blockIdx.y;
    const int b  = blockIdx.z;

    // Q fragments for d-chunks {0..31, 32..63}
    const uint16_t* Qrow =
        Qp + (size_t)(b * SEQ + q0 + lo) * EMB + h * HEAD_DIM;
    Frag aq[2];
#pragma unroll
    for (int t = 0; t < 2; ++t) {
        aq[t].q[0] = *(const u32x4*)(Qrow + t * 32 + hi * 8);
        aq[t].q[1] = *(const u32x4*)(Qrow + t * 32 + hi * 8 + 16);
    }

    // K base: per-lane key row (lo), per-lane d chunk for B-layout
    const uint16_t* Kbase =
        Kp + (size_t)(b * SEQ + lo) * EMB + h * HEAD_DIM;
    // V^T base: lane's d column, k-chunk by lane half; k0 added in loop
    const uint16_t* Vbase =
        VpT + (size_t)(h * HEAD_DIM + lo) * MROWS + b * SEQ + hi * 8;

    v8f acc[4];
#pragma unroll
    for (int t = 0; t < 4; ++t)
#pragma unroll
        for (int r = 0; r < 8; ++r) acc[t][r] = 0.0f;

    float mrun[8], lrun[8];
#pragma unroll
    for (int r = 0; r < 8; ++r) { mrun[r] = -3.0e38f; lrun[r] = 0.0f; }

    for (int k0 = 0; k0 < SEQ; k0 += 64) {
        // ---- scores: four 16x16 key subtiles, K-dim = head dim (2 chunks)
        v8f s[4];
#pragma unroll
        for (int j = 0; j < 4; ++j)
#pragma unroll
            for (int r = 0; r < 8; ++r) s[j][r] = 0.0f;
#pragma unroll
        for (int t = 0; t < 2; ++t) {
            const uint16_t* Kr =
                Kbase + (size_t)k0 * EMB + t * 32 + hi * 8;
#pragma unroll
            for (int j = 0; j < 4; ++j) {
                const uint16_t* Kj = Kr + (size_t)(j * 16) * EMB;
                Frag bk;
                bk.q[0] = *(const u32x4*)Kj;
                bk.q[1] = *(const u32x4*)(Kj + 16);
                s[j] = __builtin_amdgcn_wmma_f32_16x16x32_bf16(
                    false, aq[t].v, false, bk.v, (short)0, s[j], false, false);
            }
        }

        // ---- online softmax over these 64 keys (base-2 domain; Q prescaled)
#pragma unroll
        for (int r = 0; r < 8; ++r) {
            float v = fmaxf(fmaxf(s[0][r], s[1][r]), fmaxf(s[2][r], s[3][r]));
            v = fmaxf(v, __shfl_xor(v, 1, 32));
            v = fmaxf(v, __shfl_xor(v, 2, 32));
            v = fmaxf(v, __shfl_xor(v, 4, 32));
            v = fmaxf(v, __shfl_xor(v, 8, 32));   // stays in 16-lane half
            const float mnew  = fmaxf(mrun[r], v);
            const float alpha = exp2f(mrun[r] - mnew);
            mrun[r] = mnew;
            float rs = 0.0f;
#pragma unroll
            for (int j = 0; j < 4; ++j) {
                const float p = exp2f(s[j][r] - mnew);
                s[j][r] = p;
                rs += p;
            }
            rs += __shfl_xor(rs, 1, 32);
            rs += __shfl_xor(rs, 2, 32);
            rs += __shfl_xor(rs, 4, 32);
            rs += __shfl_xor(rs, 8, 32);
            lrun[r] = lrun[r] * alpha + rs;
#pragma unroll
            for (int t = 0; t < 4; ++t) acc[t][r] *= alpha;
        }

        // ---- stage P (C-layout f32 -> row-major bf16 in LDS, 16x64)
#pragma unroll
        for (int r = 0; r < 8; ++r) {
            const int prow = (r + 8 * hi) * 64;
#pragma unroll
            for (int j = 0; j < 4; ++j)
                Pl[prow + j * 16 + lo] = f2bf(s[j][r]);
        }
        __syncthreads();

        // ---- ctx += P * V : A-frags from LDS (two K=32 halves),
        //      B-frags contiguous from V^T
        Frag ap0, ap1;
        ap0.q[0] = *(const u32x4*)(&Pl[lo * 64 + hi * 8]);
        ap0.q[1] = *(const u32x4*)(&Pl[lo * 64 + hi * 8 + 16]);
        ap1.q[0] = *(const u32x4*)(&Pl[lo * 64 + 32 + hi * 8]);
        ap1.q[1] = *(const u32x4*)(&Pl[lo * 64 + 32 + hi * 8 + 16]);
#pragma unroll
        for (int t = 0; t < 4; ++t) {
            const uint16_t* Vp0 = Vbase + (size_t)t * 16 * MROWS + k0;
            Frag bv0, bv1;
            bv0.q[0] = *(const u32x4*)Vp0;
            bv0.q[1] = *(const u32x4*)(Vp0 + 16);
            bv1.q[0] = *(const u32x4*)(Vp0 + 32);
            bv1.q[1] = *(const u32x4*)(Vp0 + 48);
            acc[t] = __builtin_amdgcn_wmma_f32_16x16x32_bf16(
                false, ap0.v, false, bv0.v, (short)0, acc[t], false, false);
            acc[t] = __builtin_amdgcn_wmma_f32_16x16x32_bf16(
                false, ap1.v, false, bv1.v, (short)0, acc[t], false, false);
        }
        __syncthreads();   // protect Pl WAR before next iteration's stores
    }

    // ---- normalize (rcp, not div) and emit ctx (bf16, [B*S, EMB])
    float rinv[8];
#pragma unroll
    for (int r = 0; r < 8; ++r) rinv[r] = __builtin_amdgcn_rcpf(lrun[r]);
#pragma unroll
    for (int t = 0; t < 4; ++t) {
#pragma unroll
        for (int r = 0; r < 8; ++r) {
            const float val = acc[t][r] * rinv[r];
            const int row = q0 + r + 8 * hi;
            const int col = h * HEAD_DIM + t * 16 + lo;
            Ctx[(size_t)(b * SEQ + row) * EMB + col] = f2bf(val);
        }
    }
}

// ---------------------------------------------------------------------------
extern "C" void kernel_launch(void* const* d_in, const int* in_sizes, int n_in,
                              void* d_out, int out_size, void* d_ws,
                              size_t ws_size, hipStream_t stream) {
    const float* q  = (const float*)d_in[0];
    const float* k  = (const float*)d_in[1];
    const float* v  = (const float*)d_in[2];
    const float* Wq = (const float*)d_in[3];
    const float* bq = (const float*)d_in[4];
    const float* Wk = (const float*)d_in[5];
    const float* bk = (const float*)d_in[6];
    const float* Wv = (const float*)d_in[7];
    const float* bv = (const float*)d_in[8];
    const float* Wo = (const float*)d_in[9];
    const float* bo = (const float*)d_in[10];

    char* ws = (char*)d_ws;
    const size_t XB = (size_t)MROWS * EMB * 2;  // 8 MiB activation (bf16)
    const size_t WB = (size_t)EMB * EMB * 2;    // 2 MiB weight (bf16)
    uint16_t* Xq  = (uint16_t*)(ws + 0 * XB);
    uint16_t* Xk  = (uint16_t*)(ws + 1 * XB);
    uint16_t* Xv  = (uint16_t*)(ws + 2 * XB);
    uint16_t* Wqb = (uint16_t*)(ws + 3 * XB + 0 * WB);
    uint16_t* Wkb = (uint16_t*)(ws + 3 * XB + 1 * WB);
    uint16_t* Wvb = (uint16_t*)(ws + 3 * XB + 2 * WB);
    uint16_t* Wob = (uint16_t*)(ws + 3 * XB + 3 * WB);
    uint16_t* Qp  = (uint16_t*)(ws + 3 * XB + 4 * WB + 0 * XB);
    uint16_t* Kp  = (uint16_t*)(ws + 3 * XB + 4 * WB + 1 * XB);
    uint16_t* VpT = (uint16_t*)(ws + 3 * XB + 4 * WB + 2 * XB); // [EMB][M]
    uint16_t* Ctx = (uint16_t*)(ws + 3 * XB + 4 * WB + 3 * XB);
    // total workspace use: 7*8MiB + 4*2MiB = 64 MiB

    const int nAct4 = (MROWS * EMB) / 4;  // 1M float4 groups
    const int nWgt4 = (EMB * EMB) / 4;    // 256K float4 groups
    cvt_f32_bf16x4<<<(nAct4 + 255) / 256, 256, 0, stream>>>(q,  Xq,  nAct4);
    cvt_f32_bf16x4<<<(nAct4 + 255) / 256, 256, 0, stream>>>(k,  Xk,  nAct4);
    cvt_f32_bf16x4<<<(nAct4 + 255) / 256, 256, 0, stream>>>(v,  Xv,  nAct4);
    cvt_f32_bf16x4<<<(nWgt4 + 255) / 256, 256, 0, stream>>>(Wq, Wqb, nWgt4);
    cvt_f32_bf16x4<<<(nWgt4 + 255) / 256, 256, 0, stream>>>(Wk, Wkb, nWgt4);
    cvt_f32_bf16x4<<<(nWgt4 + 255) / 256, 256, 0, stream>>>(Wv, Wvb, nWgt4);
    cvt_f32_bf16x4<<<(nWgt4 + 255) / 256, 256, 0, stream>>>(Wo, Wob, nWgt4);

    // softmax scale folded into Q: 1/sqrt(64) * log2(e)
    const float qscale = 0.125f * 1.44269504f;

    const dim3 gemmGrid((MROWS / 32) * (EMB / 64));  // 2048 waves
    gemm_xwT<true,  false><<<gemmGrid, 32, 0, stream>>>(Xq, Wqb, bq, Qp,
                                                        MROWS, EMB, EMB,
                                                        qscale);
    gemm_xwT<true,  false><<<gemmGrid, 32, 0, stream>>>(Xk, Wkb, bk, Kp,
                                                        MROWS, EMB, EMB, 1.0f);
    gemm_xwT<true,  true ><<<gemmGrid, 32, 0, stream>>>(Xv, Wvb, bv, VpT,
                                                        MROWS, EMB, EMB, 1.0f);

    const dim3 attnGrid(SEQ / 16, HEADS, BATCH);     // 4096 waves
    attn_fa<<<attnGrid, 32, 0, stream>>>(Qp, Kp, VpT, Ctx);

    gemm_xwT<false, false><<<gemmGrid, 32, 0, stream>>>(Ctx, Wob, bo, d_out,
                                                        MROWS, EMB, EMB, 1.0f);
}
